// CoLA_61813169324795
// MI455X (gfx1250) — matrix operand order repaired and checked by reading, not compiled, not run on hardware
//
#include <hip/hip_runtime.h>
#include <math.h>

// ---------------------------------------------------------------------------
// CDNA5 (gfx1250, wave32) implementation of the CoLA reference pipeline.
// Heavy GEMMs (projections, attention QK^T / PV, embed conv as implicit GEMM
// with K=18432) run on v_wmma_f32_16x16x32_bf16. Blocks are 8 waves sharing
// one A tile, staged through double-buffered LDS (one barrier per 32-K chunk,
// global loads and LDS stores overlapped with the WMMAs). The dominant embed
// conv uses 4 M-tiles per wave so each B fragment feeds 4 WMMAs (4x less B
// traffic from L2). All operands are pre-relayouted to bf16 with contiguous
// K-pairs so fragment loads are dword loads (ds_load_b128 / global_load_b128).
// Padded dims (T_PAD=768) are zero-filled so K loops need no guards.
// Workspace use: ~710 MB from d_ws (bump-allocated below).
// ---------------------------------------------------------------------------

typedef __bf16 bf16_t;
typedef bf16_t v16bf __attribute__((ext_vector_type(16)));
typedef float  v8f   __attribute__((ext_vector_type(8)));

#define B_N    16
#define T_LEN  750
#define T_PAD  768
#define D_FULL 2048
#define D_HALF 1024
#define H_N    8
#define DH_N   128
#define BH_N   128     // B_N*H_N
#define C_N    20
#define K_EMB  18432   // 9*2048
#define K_EASY 150
#define K_HARD 37
#define MW     4       // M-tiles per wave in k_embed

union FragAB { v16bf v; unsigned int u[8]; unsigned short s[16]; };
union FragC  { v8f v;  float f[8]; };

__device__ __forceinline__ unsigned short f2bf(float f) {
  unsigned u = __float_as_uint(f);
  unsigned r = u + 0x7fffu + ((u >> 16) & 1u);   // round-to-nearest-even
  return (unsigned short)(r >> 16);
}
__device__ __forceinline__ float bf2f(unsigned short s) {
  return __uint_as_float(((unsigned)s) << 16);
}
__device__ __forceinline__ v8f wmma_bf16(v16bf a, v16bf b, v8f c) {
  return __builtin_amdgcn_wmma_f32_16x16x32_bf16(false, a, false, b, (short)0, c, false, false);
}
// A fragment (16x32 bf16): VGPR v holds K pair; pair index within a 16-dword row.
__device__ __forceinline__ int pairA(int v, int hi) { return v + 8 * hi; }
// B fragment (32x16 bf16): lane = column; K pair index v + 8*hi.

// ---------------------------------------------------------------------------
// Prep: elementwise f32 -> bf16
// ---------------------------------------------------------------------------
__global__ __launch_bounds__(256)
void k_cvt_bf16(const float* __restrict__ src, unsigned short* __restrict__ dst, long total)
{
  long i = (long)blockIdx.x * 256 + threadIdx.x;
  if (i < total) dst[i] = f2bf(src[i]);
}

// Prep: per-head transpose f32 [h, Kd, Nd] -> bf16 [h, Nd, Kd]
__global__ __launch_bounds__(256)
void k_wtrans(const float* __restrict__ src, unsigned short* __restrict__ dst,
              int Kd, int Nd, long total)
{
  long i = (long)blockIdx.x * 256 + threadIdx.x;
  if (i >= total) return;
  long per = (long)Kd * Nd;
  int h  = (int)(i / per);
  int r  = (int)(i % per);
  int nI = r / Kd, kI = r % Kd;
  dst[i] = f2bf(src[(size_t)h * per + (size_t)kI * Nd + nI]);
}

// Prep: transpose bf16 o [nh, m(T_PAD), e(DH)] -> oT [nh, e(DH), m(T_PAD)]
__global__ __launch_bounds__(256)
void k_otrans(const unsigned short* __restrict__ src, unsigned short* __restrict__ dst)
{
  long i = (long)blockIdx.x * 256 + threadIdx.x;
  if (i >= (long)BH_N * DH_N * T_PAD) return;
  int m  = (int)(i % T_PAD);
  int e  = (int)((i / T_PAD) % DH_N);
  int nh = (int)(i / ((long)T_PAD * DH_N));
  dst[i] = src[((size_t)nh * T_PAD + m) * DH_N + e];
}

// ---------------------------------------------------------------------------
// K1: o_pre[(n,h), l, e] = sum_d xbf[n,l,xOff+d] * pT[h,e,d]   (f32 out)
// tiles: n(16) x h(8) x lT(47) x eT(8); 8 waves/block share (n,h,lT) -> LDS A.
// Double-buffered stage: one barrier per 32-K chunk.
// ---------------------------------------------------------------------------
__global__ __launch_bounds__(256)
void k_proj(const unsigned short* __restrict__ xbf, const unsigned short* __restrict__ pT,
            float* __restrict__ o_pre, int xOff)
{
  __shared__ unsigned int As[2][256];  // 16 rows x 16 dwords, double-buffered
  int wave = blockIdx.x * 8 + (threadIdx.x >> 5);
  int eT = wave % 8;  int t1 = wave / 8;     // t1 is block-uniform
  int lT = t1 % 47;   int t2 = t1 / 47;
  int h  = t2 % H_N;  int n  = t2 / H_N;
  if (n >= B_N) return;                      // block-uniform
  int lane = threadIdx.x & 31, li = lane & 15, hi = lane >> 4;
  int col = eT * 16 + li;                    // e
  int tid = threadIdx.x, mrow = tid >> 4, cpair = tid & 15;
  int rowA = lT * 16 + mrow;
  int rowC = rowA < T_LEN ? rowA : (T_LEN - 1);   // clamp: bad rows never stored
  const unsigned short* aSrc = xbf + ((size_t)n * T_LEN + rowC) * D_FULL + xOff;
  const unsigned short* bRow = pT + ((size_t)h * DH_N + col) * D_HALF;
  FragC acc;
#pragma unroll
  for (int i = 0; i < 8; ++i) acc.f[i] = 0.f;
  As[0][tid] = *(const unsigned int*)(aSrc + cpair * 2);
  int cur = 0;
  for (int kk = 0; kk < D_HALF; kk += 32) {
    bool more = (kk + 32) < D_HALF;
    unsigned int nv = 0;
    if (more) nv = *(const unsigned int*)(aSrc + kk + 32 + cpair * 2);
    __syncthreads();
    FragAB a, b;
#pragma unroll
    for (int v = 0; v < 8; ++v) {
      a.u[v] = As[cur][li * 16 + pairA(v, hi)];
      b.u[v] = ((const unsigned int*)(bRow + kk))[v + 8 * hi];
    }
    if (more) As[cur ^ 1][tid] = nv;
    acc.v = wmma_bf16(a.v, b.v, acc.v);
    cur ^= 1;
  }
  int nh = n * H_N + h;
  float* outp = o_pre + (size_t)nh * T_PAD * DH_N;
#pragma unroll
  for (int j = 0; j < 8; ++j) {
    int r = lT * 16 + j + 8 * hi;
    if (r < T_LEN) outp[(size_t)r * DH_N + col] = acc.f[j];
  }
}

// ---------------------------------------------------------------------------
// K2: l2-normalize rows of o_pre (DH=128) -> bf16; zero the pad rows.
// ---------------------------------------------------------------------------
__global__ __launch_bounds__(128)
void k_l2norm(const float* __restrict__ o_pre, unsigned short* __restrict__ o_bf)
{
  int r = blockIdx.x;            // nh*T_PAD + l
  int t = threadIdx.x;           // 128
  int l = r % T_PAD;
  if (l >= T_LEN) { o_bf[(size_t)r * DH_N + t] = 0; return; }
  __shared__ float red[128];
  float v = o_pre[(size_t)r * DH_N + t];
  red[t] = v * v;
  __syncthreads();
  for (int s = 64; s > 0; s >>= 1) { if (t < s) red[t] += red[t + s]; __syncthreads(); }
  float scale = 1.f / fmaxf(sqrtf(red[0]), 1e-12f);
  o_bf[(size_t)r * DH_N + t] = f2bf(v * scale);
}

// ---------------------------------------------------------------------------
// K3: q[nh,l,f] = sum_e o_rgb[nh,l,e] * awT[h,f,e]  (bf16 out, pad rows = 0)
// tiles: nh(128) x lT(48) x fT(8). K=128, direct dword loads.
// ---------------------------------------------------------------------------
__global__ __launch_bounds__(256)
void k_qgemm(const unsigned short* __restrict__ o_rgb, const unsigned short* __restrict__ awT,
             unsigned short* __restrict__ q)
{
  int wave = blockIdx.x * 8 + (threadIdx.x >> 5);
  int fT = wave % 8;  int t1 = wave / 8;
  int lT = t1 % 48;   int nh = t1 / 48;
  if (nh >= BH_N) return;
  int h = nh & 7;
  int lane = threadIdx.x & 31, li = lane & 15, hi = lane >> 4;
  int row = lT * 16 + li;
  int col = fT * 16 + li;
  const unsigned short* aRow = o_rgb + ((size_t)nh * T_PAD + row) * DH_N;
  const unsigned short* bRow = awT + ((size_t)h * DH_N + col) * DH_N;
  FragC acc;
#pragma unroll
  for (int i = 0; i < 8; ++i) acc.f[i] = 0.f;
  for (int kk = 0; kk < DH_N; kk += 32) {
    FragAB a, b;
#pragma unroll
    for (int v = 0; v < 8; ++v) {
      a.u[v] = ((const unsigned int*)(aRow + kk))[pairA(v, hi)];
      b.u[v] = ((const unsigned int*)(bRow + kk))[v + 8 * hi];
    }
    acc.v = wmma_bf16(a.v, b.v, acc.v);
  }
  unsigned short* outp = q + (size_t)nh * T_PAD * DH_N;
#pragma unroll
  for (int j = 0; j < 8; ++j) {
    int r = lT * 16 + j + 8 * hi;
    outp[(size_t)r * DH_N + col] = f2bf(acc.f[j]);
  }
}

// ---------------------------------------------------------------------------
// K4: atte[nh,l,m] = sum_f q[nh,l,f] * o_flow[nh,m,f]  (bf16, 768x768 per nh)
// tiles: nh(128) x lT(48) x mT(48). Pad rows/cols come out 0 (zero inputs).
// ---------------------------------------------------------------------------
__global__ __launch_bounds__(256)
void k_attn(const unsigned short* __restrict__ q, const unsigned short* __restrict__ o_flow,
            unsigned short* __restrict__ atte)
{
  int wave = blockIdx.x * 8 + (threadIdx.x >> 5);
  int mT = wave % 48; int t1 = wave / 48;
  int lT = t1 % 48;   int nh = t1 / 48;
  if (nh >= BH_N) return;
  int lane = threadIdx.x & 31, li = lane & 15, hi = lane >> 4;
  int row  = lT * 16 + li;   // l
  int colM = mT * 16 + li;   // m
  const unsigned short* aRow = q      + ((size_t)nh * T_PAD + row)  * DH_N;
  const unsigned short* bRow = o_flow + ((size_t)nh * T_PAD + colM) * DH_N;
  FragC acc;
#pragma unroll
  for (int i = 0; i < 8; ++i) acc.f[i] = 0.f;
  for (int kk = 0; kk < DH_N; kk += 32) {
    FragAB a, b;
#pragma unroll
    for (int v = 0; v < 8; ++v) {
      a.u[v] = ((const unsigned int*)(aRow + kk))[pairA(v, hi)];
      b.u[v] = ((const unsigned int*)(bRow + kk))[v + 8 * hi];
    }
    acc.v = wmma_bf16(a.v, b.v, acc.v);
  }
  unsigned short* outp = atte + (size_t)nh * T_PAD * T_PAD;
#pragma unroll
  for (int j = 0; j < 8; ++j) {
    int r = lT * 16 + j + 8 * hi;
    outp[(size_t)r * T_PAD + colM] = f2bf(acc.f[j]);
  }
}

// ---------------------------------------------------------------------------
// K5/K7: softmax over valid m (<750) -> P (bf16), zero pad cols and pad rows.
// ---------------------------------------------------------------------------
__global__ __launch_bounds__(256)
void k_softmax_rows(const unsigned short* __restrict__ atte, unsigned short* __restrict__ P)
{
  int r = blockIdx.x;             // nh*T_PAD + l
  int l = r % T_PAD;
  int t = threadIdx.x;
  unsigned short* outp = P + (size_t)r * T_PAD;
  if (l >= T_LEN) { for (int m = t; m < T_PAD; m += 256) outp[m] = 0; return; }
  const unsigned short* in = atte + (size_t)r * T_PAD;
  __shared__ float red[256];
  float mx = -3.4e38f;
  for (int m = t; m < T_LEN; m += 256) mx = fmaxf(mx, bf2f(in[m]));
  red[t] = mx; __syncthreads();
  for (int s = 128; s > 0; s >>= 1) { if (t < s) red[t] = fmaxf(red[t], red[t + s]); __syncthreads(); }
  mx = red[0]; __syncthreads();
  float sm = 0.f;
  for (int m = t; m < T_LEN; m += 256) sm += expf(bf2f(in[m]) - mx);
  red[t] = sm; __syncthreads();
  for (int s = 128; s > 0; s >>= 1) { if (t < s) red[t] += red[t + s]; __syncthreads(); }
  float inv = 1.f / red[0];
  for (int m = t; m < T_PAD; m += 256)
    outp[m] = (m < T_LEN) ? f2bf(expf(bf2f(in[m]) - mx) * inv) : (unsigned short)0;
}

__global__ __launch_bounds__(256)
void k_softmax_cols(const unsigned short* __restrict__ atte, unsigned short* __restrict__ P)
{
  int r = blockIdx.x;             // nh*T_PAD + l
  int nh = r / T_PAD; int l = r % T_PAD;
  int t = threadIdx.x;
  unsigned short* outp = P + (size_t)r * T_PAD;
  if (l >= T_LEN) { for (int m = t; m < T_PAD; m += 256) outp[m] = 0; return; }
  const unsigned short* base = atte + (size_t)nh * T_PAD * T_PAD + l;  // elem m at base[m*T_PAD]
  __shared__ float red[256];
  float mx = -3.4e38f;
  for (int m = t; m < T_LEN; m += 256) mx = fmaxf(mx, bf2f(base[(size_t)m * T_PAD]));
  red[t] = mx; __syncthreads();
  for (int s = 128; s > 0; s >>= 1) { if (t < s) red[t] = fmaxf(red[t], red[t + s]); __syncthreads(); }
  mx = red[0]; __syncthreads();
  float sm = 0.f;
  for (int m = t; m < T_LEN; m += 256) sm += expf(bf2f(base[(size_t)m * T_PAD]) - mx);
  red[t] = sm; __syncthreads();
  for (int s = 128; s > 0; s >>= 1) { if (t < s) red[t] += red[t + s]; __syncthreads(); }
  float inv = 1.f / red[0];
  for (int m = t; m < T_PAD; m += 256)
    outp[m] = (m < T_LEN) ? f2bf(expf(bf2f(base[(size_t)m * T_PAD]) - mx) * inv) : (unsigned short)0;
}

// ---------------------------------------------------------------------------
// K6/K8: e = P @ O ; epilogue: xc[n,l,half*1024+h*128+e] =
//        bf16( tanh( gelu_exact(e) + x[...] ) )
// tiles: nh(128) x lT(47) x eT(8); K = 768 (zero-padded); double-buffered
// LDS-staged A; B from oT (contiguous K-pairs).
// ---------------------------------------------------------------------------
__global__ __launch_bounds__(256)
void k_egemm(const unsigned short* __restrict__ P, const unsigned short* __restrict__ oT,
             const float* __restrict__ x, unsigned short* __restrict__ xc, int half)
{
  __shared__ unsigned int As[2][256];
  int wave = blockIdx.x * 8 + (threadIdx.x >> 5);
  int eT = wave % 8;  int t1 = wave / 8;     // t1 block-uniform
  int lT = t1 % 47;   int nh = t1 / 47;
  if (nh >= BH_N) return;
  int n = nh >> 3, h = nh & 7;
  int lane = threadIdx.x & 31, li = lane & 15, hi = lane >> 4;
  int col = eT * 16 + li;                    // e within head
  int tid = threadIdx.x, mrow = tid >> 4, cpair = tid & 15;
  int rowA = lT * 16 + mrow;                 // <= 751 < T_PAD (zero pad rows)
  const unsigned short* aSrc = P + ((size_t)nh * T_PAD + rowA) * T_PAD;
  const unsigned short* bRow = oT + ((size_t)nh * DH_N + col) * T_PAD;
  FragC acc;
#pragma unroll
  for (int i = 0; i < 8; ++i) acc.f[i] = 0.f;
  As[0][tid] = *(const unsigned int*)(aSrc + cpair * 2);
  int cur = 0;
  for (int kk = 0; kk < T_PAD; kk += 32) {
    bool more = (kk + 32) < T_PAD;
    unsigned int nv = 0;
    if (more) nv = *(const unsigned int*)(aSrc + kk + 32 + cpair * 2);
    __syncthreads();
    FragAB a, b;
#pragma unroll
    for (int v = 0; v < 8; ++v) {
      a.u[v] = As[cur][li * 16 + pairA(v, hi)];
      b.u[v] = ((const unsigned int*)(bRow + kk))[v + 8 * hi];
    }
    if (more) As[cur ^ 1][tid] = nv;
    acc.v = wmma_bf16(a.v, b.v, acc.v);
    cur ^= 1;
  }
  int dcol = half * D_HALF + h * DH_N + col;
#pragma unroll
  for (int j = 0; j < 8; ++j) {
    int r = lT * 16 + j + 8 * hi;
    if (r < T_LEN) {
      float g   = acc.f[j];
      float gel = 0.5f * g * (1.f + erff(g * 0.70710678118654752f));
      float xin = x[((size_t)n * T_LEN + r) * D_FULL + dcol];
      xc[((size_t)n * T_LEN + r) * D_FULL + dcol] = f2bf(tanhf(gel + xin));
    }
  }
}

// ---------------------------------------------------------------------------
// K9a: relayout embed_w [co,ci,k] f32 -> wB [co, k*2048+ci] bf16 (contig K').
// ---------------------------------------------------------------------------
__global__ __launch_bounds__(256)
void k_wrelayout(const float* __restrict__ embed_w, unsigned short* __restrict__ wB)
{
  size_t idx = (size_t)blockIdx.x * 256 + threadIdx.x;
  if (idx >= (size_t)D_FULL * K_EMB) return;
  int co = (int)(idx / K_EMB);
  int kp = (int)(idx % K_EMB);
  int k = kp >> 11, ci = kp & 2047;
  wB[idx] = f2bf(embed_w[((size_t)co * D_FULL + ci) * 9 + k]);
}

// branch-free conv window load (clamped address + select)
__device__ __forceinline__ unsigned int load_xc(const unsigned short* __restrict__ xc,
                                                int n, int rowC, int kk, int cpair)
{
  int k  = kk >> 11;
  int ci = (kk & 2047) + cpair * 2;
  int tt = rowC + k - 4;
  int ttc = tt < 0 ? 0 : (tt >= T_LEN ? T_LEN - 1 : tt);
  unsigned int v = *(const unsigned int*)(xc + ((size_t)n * T_LEN + ttc) * D_FULL + ci);
  return (tt >= 0 && tt < T_LEN) ? v : 0u;
}

// ---------------------------------------------------------------------------
// K9: implicit-GEMM embed conv:
//   emb[n,t,co] = relu( bias[co] + sum_{k,ci} xc[n,t+k-4,ci]*wB[co,k*2048+ci] )
// Block = 64 rows x 128 cols: 8 waves, each wave 1 N-tile x MW(4) M-tiles so
// one B fragment feeds 4 WMMAs. Double-buffered LDS A stage (8 KB), one
// barrier per 32-K chunk. grid: n(16) x mB(12) x cG(16).
// ---------------------------------------------------------------------------
__global__ __launch_bounds__(256)
void k_embed(const unsigned short* __restrict__ xc, const unsigned short* __restrict__ wB,
             const float* __restrict__ embed_b, float* __restrict__ emb)
{
  __shared__ unsigned int As[2][64 * 16];    // 64 rows x 16 dwords, double-buffered
  int blk = blockIdx.x;
  int cG = blk % 16;  int t1 = blk / 16;
  int mB = t1 % 12;   int n  = t1 / 12;
  if (n >= B_N) return;
  int wid = threadIdx.x >> 5;
  int lane = threadIdx.x & 31, li = lane & 15, hi = lane >> 4;
  int col = cG * 128 + wid * 16 + li;          // co
  int rowBase = mB * 64;
  const unsigned short* bRow = wB + (size_t)col * K_EMB;
  int tid = threadIdx.x, r0 = tid >> 4, cpair = tid & 15;
  int rowC[MW];
#pragma unroll
  for (int r = 0; r < MW; ++r) {
    int rr = rowBase + r0 + 16 * r;
    rowC[r] = rr < T_LEN ? rr : (T_LEN - 1);   // clamp: bad rows never stored
  }
  FragC acc[MW];
#pragma unroll
  for (int w = 0; w < MW; ++w)
#pragma unroll
    for (int i = 0; i < 8; ++i) acc[w].f[i] = 0.f;
  // prologue: stage chunk 0 into buffer 0
#pragma unroll
  for (int r = 0; r < MW; ++r)
    As[0][(r0 + 16 * r) * 16 + cpair] = load_xc(xc, n, rowC[r], 0, cpair);
  int cur = 0;
  for (int kk = 0; kk < K_EMB; kk += 32) {
    bool more = (kk + 32) < K_EMB;
    unsigned int nv[MW];
    if (more) {
#pragma unroll
      for (int r = 0; r < MW; ++r) nv[r] = load_xc(xc, n, rowC[r], kk + 32, cpair);
    }
    __syncthreads();     // As[cur] visible; prior reads of As[cur^1] complete
    FragAB b;
#pragma unroll
    for (int v = 0; v < 8; ++v)
      b.u[v] = ((const unsigned int*)(bRow + kk))[v + 8 * hi];
#pragma unroll
    for (int w = 0; w < MW; ++w) {
      FragAB a;
#pragma unroll
      for (int v = 0; v < 8; ++v)
        a.u[v] = As[cur][(w * 16 + li) * 16 + pairA(v, hi)];
      acc[w].v = wmma_bf16(a.v, b.v, acc[w].v);
    }
    if (more) {
#pragma unroll
      for (int r = 0; r < MW; ++r) As[cur ^ 1][(r0 + 16 * r) * 16 + cpair] = nv[r];
    }
    cur ^= 1;
  }
  float bias = embed_b[col];
#pragma unroll
  for (int w = 0; w < MW; ++w) {
#pragma unroll
    for (int j = 0; j < 8; ++j) {
      int r = rowBase + w * 16 + j + 8 * hi;
      if (r < T_LEN)
        emb[((size_t)n * T_LEN + r) * D_FULL + col] = fmaxf(acc[w].f[j] + bias, 0.f);
    }
  }
}

// ---------------------------------------------------------------------------
// K10a: cls1 conv (scalar): out1[n,c,t] = sum_{k,ci} emb[n,t+k-4,ci]*w[c,ci,k]
// ---------------------------------------------------------------------------
__global__ __launch_bounds__(256)
void k_cls1(const float* __restrict__ emb, const float* __restrict__ cls1_w,
            float* __restrict__ out1)
{
  int idx = blockIdx.x * 256 + threadIdx.x;
  if (idx >= B_N * C_N * T_LEN) return;
  int t = idx % T_LEN; int c = (idx / T_LEN) % C_N; int n = idx / (T_LEN * C_N);
  float acc = 0.f;
  for (int k = 0; k < 9; ++k) {
    int tt = t + k - 4;
    if (tt < 0 || tt >= T_LEN) continue;
    const float* er = emb + ((size_t)n * T_LEN + tt) * D_FULL;
    const float* wr = cls1_w + (size_t)c * D_FULL * 9 + k;
    for (int ci = 0; ci < D_FULL; ++ci) acc += er[ci] * wr[(size_t)ci * 9];
  }
  out1[((size_t)n * C_N + c) * T_LEN + t] = acc;
}

// K10b: cls2 depthwise (k=7, dil=2, pad=6) + relu; cas stored [n,t,c].
__global__ __launch_bounds__(256)
void k_cls2(const float* __restrict__ out1, const float* __restrict__ cls2_w,
            float* __restrict__ cas)
{
  int idx = blockIdx.x * 256 + threadIdx.x;
  if (idx >= B_N * C_N * T_LEN) return;
  int t = idx % T_LEN; int c = (idx / T_LEN) % C_N; int n = idx / (T_LEN * C_N);
  float acc = 0.f;
  for (int j = 0; j < 7; ++j) {
    int tt = t + 2 * j - 6;
    if (tt < 0 || tt >= T_LEN) continue;
    acc += out1[((size_t)n * C_N + c) * T_LEN + tt] * cls2_w[c * 7 + j];
  }
  cas[((size_t)n * T_LEN + t) * C_N + c] = fmaxf(acc, 0.f);
}

// K10c: actionness[n,t] = sum_c cas[n,t,c]
__global__ __launch_bounds__(256)
void k_actionness(const float* __restrict__ cas, float* __restrict__ act)
{
  int idx = blockIdx.x * 256 + threadIdx.x;
  if (idx >= B_N * T_LEN) return;
  const float* p = cas + (size_t)idx * C_N;
  float s = 0.f;
  for (int c = 0; c < C_N; ++c) s += p[c];
  act[idx] = s;
}

// ---------------------------------------------------------------------------
// K11: per-n median / binarize / erode-dilate (reference's zero-padding
// windows) / a_rev.  One block per n.
// ---------------------------------------------------------------------------
__global__ __launch_bounds__(256)
void k_morpho(const float* __restrict__ act, float* __restrict__ arev,
              float* __restrict__ inner, float* __restrict__ outer)
{
  __shared__ float a[T_LEN];
  __shared__ float bin[T_LEN];
  __shared__ float red[256];
  __shared__ float m0, m1;
  int n = blockIdx.x, t = threadIdx.x;
  for (int i = t; i < T_LEN; i += 256) a[i] = act[n * T_LEN + i];
  __syncthreads();
  float mx = -3.4e38f;
  for (int i = t; i < T_LEN; i += 256) mx = fmaxf(mx, a[i]);
  red[t] = mx; __syncthreads();
  for (int s = 128; s > 0; s >>= 1) { if (t < s) red[t] = fmaxf(red[t], red[t + s]); __syncthreads(); }
  float amax = red[0]; __syncthreads();
  // order statistics 374 & 375 (0-indexed) of 750 -> median
  for (int i = t; i < T_LEN; i += 256) {
    float vi = a[i]; int cl = 0, ce = 0;
    for (int s2 = 0; s2 < T_LEN; ++s2) { float vs = a[s2]; cl += (vs < vi); ce += (vs == vi); }
    if (cl <= 374 && 374 < cl + ce) m0 = vi;
    if (cl <= 375 && 375 < cl + ce) m1 = vi;
  }
  __syncthreads();
  float med = 0.5f * (m0 + m1);
  for (int i = t; i < T_LEN; i += 256) bin[i] = (a[i] > med) ? 1.f : 0.f;
  __syncthreads();
  for (int i = t; i < T_LEN; i += 256) {
    float e3 = 1e30f, e6 = 1e30f, d6 = -1e30f, d3 = -1e30f;
    for (int j = i - 1; j <= i + 1; ++j) {          // erode w=3: [t-1,t+1]
      float b = (j >= 0 && j < T_LEN) ? bin[j] : 0.f; e3 = fminf(e3, b);
    }
    for (int j = i - 3; j <= i + 2; ++j) {          // erode w=6: [t-3,t+2]
      float b = (j >= 0 && j < T_LEN) ? bin[j] : 0.f; e6 = fminf(e6, b);
    }
    for (int j = i - 2; j <= i + 3; ++j) {          // dilate w=6: [t-2,t+3]
      float b = (j >= 0 && j < T_LEN) ? bin[j] : 0.f; d6 = fmaxf(d6, b);
    }
    for (int j = i - 1; j <= i + 1; ++j) {          // dilate w=3: [t-1,t+1]
      float b = (j >= 0 && j < T_LEN) ? bin[j] : 0.f; d3 = fmaxf(d3, b);
    }
    inner[n * T_LEN + i] = a[i] * (e3 - e6);
    outer[n * T_LEN + i] = a[i] * (d6 - d3);
    arev [n * T_LEN + i] = amax - a[i];
  }
}

// ---------------------------------------------------------------------------
// K12: stable descending top-k index selection by rank counting (matches
// lax.top_k tie order). One block per n.
// ---------------------------------------------------------------------------
__global__ __launch_bounds__(256)
void k_select(const float* __restrict__ score, int k, int* __restrict__ idxOut)
{
  __shared__ float s[T_LEN];
  int n = blockIdx.x, t = threadIdx.x;
  for (int i = t; i < T_LEN; i += 256) s[i] = score[n * T_LEN + i];
  __syncthreads();
  for (int i = t; i < T_LEN; i += 256) {
    float vi = s[i]; int r = 0;
    for (int j = 0; j < T_LEN; ++j) {
      float vj = s[j];
      r += (vj > vi) || (vj == vi && j < i);
    }
    if (r < k) idxOut[n * k + r] = i;
  }
}

// K13: gather embedding rows into output.
__global__ __launch_bounds__(256)
void k_gather(const float* __restrict__ emb, const int* __restrict__ idx, int k,
              float* __restrict__ outp)
{
  int b = blockIdx.x;             // n*k + slot
  int n = b / k;
  int t = idx[b];
  const float* src = emb + ((size_t)n * T_LEN + t) * D_FULL;
  float* dst = outp + (size_t)b * D_FULL;
  for (int i = threadIdx.x; i < D_FULL; i += 256) dst[i] = src[i];
}

// K14: mean of top-150 cas values per (n,c).
__global__ __launch_bounds__(256)
void k_topcas(const float* __restrict__ cas, float* __restrict__ vmean)
{
  __shared__ float s[T_LEN];
  __shared__ float red[256];
  int b = blockIdx.x;             // n*C_N + c
  int n = b / C_N, c = b % C_N;
  for (int i = threadIdx.x; i < T_LEN; i += 256)
    s[i] = cas[((size_t)n * T_LEN + i) * C_N + c];
  __syncthreads();
  float local = 0.f;
  for (int i = threadIdx.x; i < T_LEN; i += 256) {
    float vi = s[i]; int r = 0;
    for (int j = 0; j < T_LEN; ++j) { float vj = s[j]; r += (vj > vi) || (vj == vi && j < i); }
    if (r < K_EASY) local += vi;
  }
  red[threadIdx.x] = local; __syncthreads();
  for (int st = 128; st > 0; st >>= 1) {
    if (threadIdx.x < st) red[threadIdx.x] += red[threadIdx.x + st];
    __syncthreads();
  }
  if (threadIdx.x == 0) vmean[b] = red[0] / (float)K_EASY;
}

// K15: softmax over classes -> video_scores.
__global__ void k_vscore(const float* __restrict__ vmean, float* __restrict__ outp)
{
  int n = blockIdx.x;
  if (threadIdx.x == 0) {
    float mx = -3.4e38f;
    for (int c = 0; c < C_N; ++c) mx = fmaxf(mx, vmean[n * C_N + c]);
    float sm = 0.f;
    for (int c = 0; c < C_N; ++c) sm += expf(vmean[n * C_N + c] - mx);
    for (int c = 0; c < C_N; ++c) outp[n * C_N + c] = expf(vmean[n * C_N + c] - mx) / sm;
  }
}

// ---------------------------------------------------------------------------
// Host launcher
// ---------------------------------------------------------------------------
extern "C" void kernel_launch(void* const* d_in, const int* in_sizes, int n_in,
                              void* d_out, int out_size, void* d_ws, size_t ws_size,
                              hipStream_t stream)
{
  (void)in_sizes; (void)n_in; (void)out_size; (void)ws_size;
  const float* x         = (const float*)d_in[0];
  const float* rgb_proj  = (const float*)d_in[1];
  const float* flow_proj = (const float*)d_in[2];
  const float* atte_w    = (const float*)d_in[3];
  const float* embed_w   = (const float*)d_in[4];
  const float* embed_b   = (const float*)d_in[5];
  const float* cls1_w    = (const float*)d_in[6];
  const float* cls2_w    = (const float*)d_in[7];

  char* wsb = (char*)d_ws;
  size_t off = 0;
  auto take = [&](size_t bytes) -> void* {
    void* p = (void*)(wsb + off);
    off = (off + bytes + 255u) & ~((size_t)255u);
    return p;
  };
  float*          o_pre = (float*)         take((size_t)BH_N * T_PAD * DH_N * 4);   //  50 MB
  unsigned short* o_rgb = (unsigned short*)take((size_t)BH_N * T_PAD * DH_N * 2);   //  25 MB
  unsigned short* o_flw = (unsigned short*)take((size_t)BH_N * T_PAD * DH_N * 2);   //  25 MB
  unsigned short* qbuf  = (unsigned short*)take((size_t)BH_N * T_PAD * DH_N * 2);   //  25 MB
  unsigned short* atte  = (unsigned short*)take((size_t)BH_N * T_PAD * T_PAD * 2);  // 151 MB
  unsigned short* Pbuf  = (unsigned short*)take((size_t)BH_N * T_PAD * T_PAD * 2);  // 151 MB
  unsigned short* xc    = (unsigned short*)take((size_t)B_N * T_LEN * D_FULL * 2);  //  49 MB
  unsigned short* wB    = (unsigned short*)take((size_t)D_FULL * K_EMB * 2);        //  75 MB
  float*          emb   = (float*)         take((size_t)B_N * T_LEN * D_FULL * 4);  //  98 MB
  unsigned short* xbf   = (unsigned short*)take((size_t)B_N * T_LEN * D_FULL * 2);  //  49 MB
  unsigned short* pT    = (unsigned short*)take((size_t)H_N * DH_N * D_HALF * 2);   //   2 MB
  unsigned short* awT   = (unsigned short*)take((size_t)H_N * DH_N * DH_N * 2);
  unsigned short* oT    = (unsigned short*)take((size_t)BH_N * DH_N * T_PAD * 2);   //  25 MB
  float*          out1  = (float*)         take((size_t)B_N * C_N * T_LEN * 4);
  float*          arev  = (float*)         take((size_t)B_N * T_LEN * 4);
  float*          inner = (float*)         take((size_t)B_N * T_LEN * 4);
  float*          outer = (float*)         take((size_t)B_N * T_LEN * 4);
  float*          vmean = (float*)         take((size_t)B_N * C_N * 4);
  int*            idxEA = (int*)           take((size_t)B_N * K_EASY * 4);
  int*            idxEB = (int*)           take((size_t)B_N * K_EASY * 4);
  int*            idxHA = (int*)           take((size_t)B_N * K_HARD * 4);
  int*            idxHB = (int*)           take((size_t)B_N * K_HARD * 4);

  float* out     = (float*)d_out;
  float* out_vs  = out;
  float* out_ea  = out + (size_t)B_N * C_N;
  float* out_eb  = out_ea + (size_t)B_N * K_EASY * D_FULL;
  float* out_ha  = out_eb + (size_t)B_N * K_EASY * D_FULL;
  float* out_hb  = out_ha + (size_t)B_N * K_HARD * D_FULL;
  float* out_act = out_hb + (size_t)B_N * K_HARD * D_FULL;
  float* out_cas = out_act + (size_t)B_N * T_LEN;

  // 0) operand relayouts (bf16, contiguous K-pairs)
  k_cvt_bf16<<<96000, 256, 0, stream>>>(x, xbf, (long)B_N * T_LEN * D_FULL);
  // 1) projections + l2norm (rgb then flow; o_pre and pT reused)
  k_wtrans<<<4096, 256, 0, stream>>>(rgb_proj, pT, D_HALF, DH_N, (long)H_N * D_HALF * DH_N);
  k_proj  <<<6016, 256, 0, stream>>>(xbf, pT, o_pre, 0);
  k_l2norm<<<BH_N * T_PAD, 128, 0, stream>>>(o_pre, o_rgb);
  k_wtrans<<<4096, 256, 0, stream>>>(flow_proj, pT, D_HALF, DH_N, (long)H_N * D_HALF * DH_N);
  k_proj  <<<6016, 256, 0, stream>>>(xbf, pT, o_pre, D_HALF);
  k_l2norm<<<BH_N * T_PAD, 128, 0, stream>>>(o_pre, o_flw);
  // 2) attention
  k_wtrans<<<512, 256, 0, stream>>>(atte_w, awT, DH_N, DH_N, (long)H_N * DH_N * DH_N);
  k_qgemm<<<6144, 256, 0, stream>>>(o_rgb, awT, qbuf);
  k_attn <<<36864, 256, 0, stream>>>(qbuf, o_flw, atte);
  k_softmax_rows<<<BH_N * T_PAD, 256, 0, stream>>>(atte, Pbuf);
  k_otrans<<<49152, 256, 0, stream>>>(o_rgb, oT);
  k_egemm<<<6016, 256, 0, stream>>>(Pbuf, oT, x, xc, 0);
  k_softmax_cols<<<BH_N * T_PAD, 256, 0, stream>>>(atte, Pbuf);
  k_otrans<<<49152, 256, 0, stream>>>(o_flw, oT);
  k_egemm<<<6016, 256, 0, stream>>>(Pbuf, oT, x, xc, 1);
  // 3) embed conv (implicit GEMM, K=18432, 4 M-tiles/wave, dbl-buffered LDS)
  k_wrelayout<<<147456, 256, 0, stream>>>(embed_w, wB);
  k_embed<<<3072, 256, 0, stream>>>(xc, wB, embed_b, emb);
  // 4) classifier convs + actionness
  k_cls1<<<(B_N * C_N * T_LEN + 255) / 256, 256, 0, stream>>>(emb, cls1_w, out1);
  k_cls2<<<(B_N * C_N * T_LEN + 255) / 256, 256, 0, stream>>>(out1, cls2_w, out_cas);
  k_actionness<<<(B_N * T_LEN + 255) / 256, 256, 0, stream>>>(out_cas, out_act);
  // 5) median / morphology / reversed scores
  k_morpho<<<B_N, 256, 0, stream>>>(out_act, arev, inner, outer);
  // 6) top-k selections + gathers
  k_select<<<B_N, 256, 0, stream>>>(out_act, K_EASY, idxEA);
  k_select<<<B_N, 256, 0, stream>>>(arev,    K_EASY, idxEB);
  k_select<<<B_N, 256, 0, stream>>>(inner,   K_HARD, idxHA);
  k_select<<<B_N, 256, 0, stream>>>(outer,   K_HARD, idxHB);
  k_gather<<<B_N * K_EASY, 256, 0, stream>>>(emb, idxEA, K_EASY, out_ea);
  k_gather<<<B_N * K_EASY, 256, 0, stream>>>(emb, idxEB, K_EASY, out_eb);
  k_gather<<<B_N * K_HARD, 256, 0, stream>>>(emb, idxHA, K_HARD, out_ha);
  k_gather<<<B_N * K_HARD, 256, 0, stream>>>(emb, idxHB, K_HARD, out_hb);
  // 7) video scores
  k_topcas<<<B_N * C_N, 256, 0, stream>>>(out_cas, vmean);
  k_vscore<<<B_N, 32, 0, stream>>>(vmean, out_vs);
}
